// SimpleTransition_48301202210919
// MI455X (gfx1250) — matrix operand order repaired and checked by reading, not compiled, not run on hardware
//
#include <hip/hip_runtime.h>

#define B_SZ    4
#define SEQ_IN  4096
#define SEQ_OUT 2048
#define DIN     2048
#define DOUT    1024
#define EPSF    1e-5f
#define HSTRIDE 1032   // 1024 + 8 ushort pad -> conflict-free ds_load_b128 across 16 rows

typedef __attribute__((ext_vector_type(16))) __bf16          v16bf;
typedef __attribute__((ext_vector_type(8)))  float           v8f;
typedef __attribute__((ext_vector_type(8)))  unsigned short  v8us;

union Frag { v16bf bf; v8us h[2]; };

__device__ __forceinline__ unsigned short f2bf(float f) {
  unsigned u = __float_as_uint(f);
  u += 0x7FFFu + ((u >> 16) & 1u);   // round-to-nearest-even
  return (unsigned short)(u >> 16);
}

// ---------------------------------------------------------------------------
// Stable compaction of True mask positions: one 256-thread block per batch.
// Thread t owns contiguous chunk [t*16, t*16+16) -> stable order preserved.
// ---------------------------------------------------------------------------
__global__ void scan_mask_kernel(const int* __restrict__ mask,
                                 int* __restrict__ idx, int* __restrict__ cnt) {
  const int b = blockIdx.x;
  const int t = threadIdx.x;                 // 256 threads
  const int* m = mask + b * SEQ_IN;
  int* idxb = idx + b * SEQ_OUT;

  int local = 0;
  #pragma unroll
  for (int i = 0; i < 16; ++i) local += (m[t * 16 + i] != 0);

  __shared__ int sc[256];
  sc[t] = local;
  __syncthreads();
  for (int off = 1; off < 256; off <<= 1) {
    int v = (t >= off) ? sc[t - off] : 0;
    __syncthreads();
    sc[t] += v;
    __syncthreads();
  }
  int k = sc[t] - local;                     // exclusive prefix
  #pragma unroll
  for (int i = 0; i < 16; ++i) {
    if (m[t * 16 + i] != 0) {
      if (k < SEQ_OUT) idxb[k] = t * 16 + i;
      ++k;
    }
  }
  if (t == 255) cnt[b] = (sc[255] < SEQ_OUT) ? sc[255] : SEQ_OUT;
}

// ---------------------------------------------------------------------------
// Elementwise f32 -> bf16 (weight layouts are already K-major per WMMA column)
// ---------------------------------------------------------------------------
__global__ void cvt_bf16_kernel(const float* __restrict__ src,
                                unsigned short* __restrict__ dst, int n) {
  int i = blockIdx.x * blockDim.x + threadIdx.x;
  if (i < n) dst[i] = f2bf(src[i]);
}

// ---------------------------------------------------------------------------
// Gather selected x rows, RMSNorm (D=2048) * g_down, emit bf16 rows.
// One 256-thread block per output row; invalid rows -> zeros.
// ---------------------------------------------------------------------------
__global__ void gather_norm_kernel(const float* __restrict__ x,
                                   const float* __restrict__ g_down,
                                   const int* __restrict__ idx,
                                   const int* __restrict__ cnt,
                                   unsigned short* __restrict__ Xg) {
  const int row = blockIdx.x;                 // 0 .. 4*2048-1
  const int b = row >> 11, j = row & 2047;
  const int t = threadIdx.x;                  // 256
  unsigned short* dst = Xg + (size_t)row * DIN;

  if (j >= cnt[b]) {                          // uniform per block
    for (int c = t; c < DIN; c += 256) dst[c] = 0;
    return;
  }
  const float* src = x + ((size_t)b * SEQ_IN + idx[b * SEQ_OUT + j]) * DIN;

  float vals[8], ss = 0.f;
  #pragma unroll
  for (int i = 0; i < 8; ++i) {
    float v = src[t + i * 256];
    vals[i] = v;
    ss += v * v;
  }
  #pragma unroll
  for (int m2 = 1; m2 < 32; m2 <<= 1) ss += __shfl_xor(ss, m2, 32);
  __shared__ float wsum[8];
  if ((t & 31) == 0) wsum[t >> 5] = ss;
  __syncthreads();
  float total = 0.f;
  #pragma unroll
  for (int w = 0; w < 8; ++w) total += wsum[w];
  float scale = rsqrtf(total * (1.0f / DIN) + EPSF);

  #pragma unroll
  for (int i = 0; i < 8; ++i) {
    int c = t + i * 256;
    dst[c] = f2bf(vals[i] * scale * g_down[c]);
  }
}

// ---------------------------------------------------------------------------
// Fused: GEMM1 (bf16 WMMA, f32 acc) -> RMSNorm*g_up -> GEMM2 -> scatter.
// Block = 16 rows, 512 threads = 16 waves.
//   GEMM1: wave covers 64 cols (4 tiles) of DOUT=1024, K=2048.
//   GEMM2: wave covers 128 cols (8 tiles) of DIN=2048, K=1024, A from LDS.
// ---------------------------------------------------------------------------
__global__ void __launch_bounds__(512)
fused_mlp_kernel(const unsigned short* __restrict__ Xg,
                 const unsigned short* __restrict__ Wd,   // [DOUT][DIN] bf16
                 const unsigned short* __restrict__ Wu,   // [DIN][DOUT] bf16
                 const float* __restrict__ g_up,
                 const int* __restrict__ idx,
                 const int* __restrict__ cnt,
                 float* __restrict__ out) {
  __shared__ unsigned short Hbf[16 * HSTRIDE];
  __shared__ float rowsq[16];

  const int blk  = blockIdx.x;               // 0..511
  const int b    = blk >> 7;
  const int j0   = (blk & 127) << 4;
  const int tid  = threadIdx.x;
  const int wave = tid >> 5;                 // 0..15
  const int lane = tid & 31;
  const int lh   = lane >> 4;                // which K-half this lane holds
  const int ln   = lane & 15;

  if (tid < 16) rowsq[tid] = 0.f;
  __syncthreads();

  // ---------------- GEMM1: H[16 x 1024] ----------------
  const unsigned short* Arow = Xg + ((size_t)(b * SEQ_OUT + j0) + ln) * DIN;
  v8f acc1[4] = {};
  for (int kk = 0; kk < DIN; kk += 32) {
    Frag a;
    a.h[0] = *(const v8us*)(Arow + kk + lh * 16);       // K = kk+16*lh+{0..7}
    a.h[1] = *(const v8us*)(Arow + kk + lh * 16 + 8);   // K = kk+16*lh+8+{0..7}
    #pragma unroll
    for (int t = 0; t < 4; ++t) {
      const int n = wave * 64 + t * 16 + ln;
      const unsigned short* bp = Wd + (size_t)n * DIN + kk + lh * 16;
      Frag bf;
      bf.h[0] = *(const v8us*)(bp);
      bf.h[1] = *(const v8us*)(bp + 8);
      acc1[t] = __builtin_amdgcn_wmma_f32_16x16x32_bf16(
          false, a.bf, false, bf.bf, (short)0, acc1[t], false, false);
    }
  }

  // ---------------- row sum-of-squares (shfl within 16-lane halves + LDS) ---
  float s[8];
  #pragma unroll
  for (int r = 0; r < 8; ++r) {
    float v = 0.f;
    #pragma unroll
    for (int t = 0; t < 4; ++t) v += acc1[t][r] * acc1[t][r];
    #pragma unroll
    for (int m2 = 1; m2 <= 8; m2 <<= 1) v += __shfl_xor(v, m2, 32);
    s[r] = v;
  }
  if (ln == 0) {
    #pragma unroll
    for (int r = 0; r < 8; ++r) atomicAdd(&rowsq[r + lh * 8], s[r]);
  }
  __syncthreads();

  // ---------------- RMSNorm * g_up -> bf16 H in LDS ----------------
  #pragma unroll
  for (int t = 0; t < 4; ++t) {
    const int n = wave * 64 + t * 16 + ln;
    const float gu = g_up[n];
    #pragma unroll
    for (int r = 0; r < 8; ++r) {
      const int M = r + lh * 8;
      const float scale = rsqrtf(rowsq[M] * (1.0f / DOUT) + EPSF);
      Hbf[M * HSTRIDE + n] = f2bf(acc1[t][r] * scale * gu);
    }
  }
  __syncthreads();

  // ---------------- GEMM2: OUT[16 x 2048], A from LDS ----------------
  v8f acc2[8] = {};
  const unsigned short* Hrow = &Hbf[ln * HSTRIDE];
  for (int kk = 0; kk < DOUT; kk += 32) {
    Frag a;
    a.h[0] = *(const v8us*)(Hrow + kk + lh * 16);
    a.h[1] = *(const v8us*)(Hrow + kk + lh * 16 + 8);
    #pragma unroll
    for (int t = 0; t < 8; ++t) {
      const int n = wave * 128 + t * 16 + ln;
      const unsigned short* bp = Wu + (size_t)n * DOUT + kk + lh * 16;
      Frag bf;
      bf.h[0] = *(const v8us*)(bp);
      bf.h[1] = *(const v8us*)(bp + 8);
      acc2[t] = __builtin_amdgcn_wmma_f32_16x16x32_bf16(
          false, a.bf, false, bf.bf, (short)0, acc2[t], false, false);
    }
  }

  // ---------------- guarded scatter ----------------
  const int K = cnt[b];
  #pragma unroll
  for (int r = 0; r < 8; ++r) {
    const int M = r + lh * 8;
    const int j = j0 + M;
    if (j < K) {
      const int tgt = idx[b * SEQ_OUT + j];
      float* orow = out + ((size_t)b * SEQ_IN + tgt) * DIN;
      #pragma unroll
      for (int t = 0; t < 8; ++t)
        orow[wave * 128 + t * 16 + ln] = acc2[t][r];
    }
  }
}

// ---------------------------------------------------------------------------
extern "C" void kernel_launch(void* const* d_in, const int* in_sizes, int n_in,
                              void* d_out, int out_size, void* d_ws, size_t ws_size,
                              hipStream_t stream) {
  (void)in_sizes; (void)n_in; (void)ws_size;
  const float* x      = (const float*)d_in[0];
  /* d_in[1] = res (only defines output shape; unused) */
  const int*   mask   = (const int*)d_in[2];
  const float* w_down = (const float*)d_in[3];
  const float* w_up   = (const float*)d_in[4];
  const float* g_down = (const float*)d_in[5];
  const float* g_up   = (const float*)d_in[6];
  float* out = (float*)d_out;

  char* ws = (char*)d_ws;
  int* idx = (int*)ws;                                       // 4*2048*4   = 32768 B
  int* cnt = (int*)(ws + 32768);                             // 16 B (padded)
  unsigned short* Xg = (unsigned short*)(ws + 33024);        // 33,554,432 B
  unsigned short* Wd = (unsigned short*)(ws + 33024 + 33554432);       // 4,194,304 B
  unsigned short* Wu = Wd + (size_t)DIN * DOUT;                        // 4,194,304 B

  hipMemsetAsync(d_out, 0, (size_t)out_size * sizeof(float), stream);

  scan_mask_kernel<<<B_SZ, 256, 0, stream>>>(mask, idx, cnt);

  const int wElems = DIN * DOUT;                             // 2,097,152
  cvt_bf16_kernel<<<(wElems + 255) / 256, 256, 0, stream>>>(w_down, Wd, wElems);
  cvt_bf16_kernel<<<(wElems + 255) / 256, 256, 0, stream>>>(w_up,   Wu, wElems);

  gather_norm_kernel<<<B_SZ * SEQ_OUT, 256, 0, stream>>>(x, g_down, idx, cnt, Xg);

  fused_mlp_kernel<<<B_SZ * (SEQ_OUT / 16), 512, 0, stream>>>(
      Xg, Wd, Wu, g_up, idx, cnt, out);
}
